// MelLinEncoder_28303834481117
// MI455X (gfx1250) — compile-verified, matrix operand order checked by reading, not run on hardware
//
#include <hip/hip_runtime.h>
#include <hip/hip_bf16.h>

// MI455X (gfx1250) implementation of the Tacotron-style MelLinEncoder.
//  - Weights converted once to bf16 ([Wih|Whh] packed per LSTM), L2-resident.
//  - LSTM gate GEMMs: WMMA bf16 (16x16x32) with f32 accumulate; the shared
//    64xK activation matrix is staged into LDS with CDNA5 async loads
//    (GLOBAL_LOAD_ASYNC_TO_LDS_B128 + s_wait_asynccnt), double-buffered, and
//    the fragment loads are software-pipelined one K-step ahead of the WMMAs.
//  - Attention is VALU (<10% of FLOPs); prenet/memory projection are WMMA.

typedef __bf16 bf16;
typedef __attribute__((ext_vector_type(16))) __bf16 v16bf;
typedef __attribute__((ext_vector_type(8)))  float  v8f;

#define B_     64
#define T_IN_  256
#define T_MEL_ 512
#define N_MEL_ 80
#define FC_    256
#define RNN_   1024
#define EMB_   512
#define ATT_   128
#define NF_    32
#define KC_    31

#define CHUNK_ 128   // K-chunk staged into LDS per double-buffer slot
#define CH_LD_ 136   // padded LDS row length (272B) -> bank-conflict-free rows

// ---------------------------------------------------------------------------
// WMMA fragment loader (global). 16-bit A/B layout: lane -> row (lane&15);
// two 16-byte groups per lane at k-offsets (lane>>4)*8 and +16.
// ---------------------------------------------------------------------------
__device__ __forceinline__ v16bf frag_ld(const bf16* __restrict__ base, int ld,
                                         int row0, int k0) {
    const int lane = threadIdx.x & 31;
    const bf16* p = base + (size_t)(row0 + (lane & 15)) * ld + k0 + ((lane >> 4) << 3);
    union { float4 f[2]; v16bf v; } u;
    u.f[0] = *(const float4*)p;
    u.f[1] = *(const float4*)(p + 16);
    return u.v;
}

// Same, but for the LDS-staged A chunk (row stride CH_LD_).
__device__ __forceinline__ v16bf frag_ld_lds(const bf16* base, int row0, int kk) {
    const int lane = threadIdx.x & 31;
    const bf16* p = base + (row0 + (lane & 15)) * CH_LD_ + kk + ((lane >> 4) << 3);
    union { float4 f[2]; v16bf v; } u;
    u.f[0] = *(const float4*)p;
    u.f[1] = *(const float4*)(p + 16);
    return u.v;
}

// ---------------------------------------------------------------------------
// Generic GEMM: Out[M,N] = act(A[M,K] @ W[N,K]^T)  (one-time prep kernels)
// ---------------------------------------------------------------------------
template <bool OUT_BF16, bool RELU>
__global__ void wmma_gemm_kernel(const bf16* __restrict__ A, int lda,
                                 const bf16* __restrict__ W, int ldw,
                                 void* __restrict__ Out, int ldo, int K) {
    const int m0   = blockIdx.x * 64;
    const int lane = threadIdx.x & 31;
    const int n0   = blockIdx.y * 128 + (threadIdx.x >> 5) * 16;
    v8f acc[4] = {};
    for (int k0 = 0; k0 < K; k0 += 32) {
        v16bf bfr = frag_ld(W, ldw, n0, k0);
#pragma unroll
        for (int mt = 0; mt < 4; ++mt) {
            v16bf afr = frag_ld(A, lda, m0 + mt * 16, k0);
            acc[mt] = __builtin_amdgcn_wmma_f32_16x16x32_bf16(
                false, afr, false, bfr, (short)0, acc[mt], false, false);
        }
    }
    const int nc = n0 + (lane & 15);
    const int mb = (lane >> 4) * 8;
#pragma unroll
    for (int mt = 0; mt < 4; ++mt) {
#pragma unroll
        for (int r = 0; r < 8; ++r) {
            const size_t m = (size_t)(m0 + mt * 16 + mb + r);
            float v = acc[mt][r];
            if (RELU) v = fmaxf(v, 0.0f);
            if (OUT_BF16) ((bf16*)Out)[m * ldo + nc] = (bf16)v;
            else          ((float*)Out)[m * ldo + nc] = v;
        }
    }
}

// ---------------------------------------------------------------------------
// LSTM step: gates[64,4096] = A[64,K] @ W[4096,K]^T + bias, fused cell update.
// Grid = 16 blocks (64 hidden units each), block = 512 threads = 16 waves.
// A is staged into LDS in 64x128 chunks with async loads (double-buffered);
// fragment loads for K-step kk+32 are issued before the WMMAs for kk.
// ---------------------------------------------------------------------------
__global__ void lstm_step_kernel(const bf16* __restrict__ A,     // [64,K]
                                 const bf16* __restrict__ W,     // [4096,K]
                                 const float* __restrict__ bias, // [4096]
                                 float* __restrict__ h_out,      // [64,1024]
                                 float* __restrict__ c_state,    // [64,1024]
                                 float* __restrict__ h_out2, long h2_stride,
                                 int K) {
    __shared__ float g[64 * 256];                       // 64 KB gate exchange
    __shared__ __align__(16) bf16 sA[2][64 * CH_LD_];   // 2 x 17 KB A chunks
    const int j0   = blockIdx.x * 64;
    const int wave = threadIdx.x >> 5;
    const int lane = threadIdx.x & 31;
    const int gate = wave >> 2;
    const int n0   = gate * RNN_ + j0 + (wave & 3) * 16;
    const int tid  = threadIdx.x;

    // Stage one 64xCHUNK_ chunk of A into sA[buf] with async loads to LDS.
    // 1024 granules of 16B; 512 threads issue 2 each. Tracked by ASYNCcnt.
    auto stage = [&](int buf, int k0) {
        for (int gi = tid; gi < 1024; gi += 512) {
            const int r  = gi >> 4;        // row 0..63
            const int gc = gi & 15;        // 16B granule within row
            const bf16* gp = A + (size_t)r * K + k0 + gc * 8;
            const unsigned lo =
                (unsigned)(size_t)(&sA[buf][r * CH_LD_]) + (unsigned)(gc * 16);
            asm volatile("global_load_async_to_lds_b128 %0, %1, off"
                         :: "v"(lo), "v"(gp) : "memory");
        }
    };

    v8f acc[4] = {};
    const int nchunks = K / CHUNK_;
    stage(0, 0);
    for (int ch = 0; ch < nchunks; ++ch) {
        asm volatile("s_wait_asynccnt 0x0" ::: "memory");
        __syncthreads();
        if (ch + 1 < nchunks) stage((ch + 1) & 1, (ch + 1) * CHUNK_);
        const bf16* Abuf = &sA[ch & 1][0];
        const int kbase = ch * CHUNK_;

        // Software-pipelined fragment loop: fetch kk+32 while computing kk.
        v16bf a_cur[4], a_nxt[4], b_cur, b_nxt;
        b_cur = frag_ld(W, K, n0, kbase);
#pragma unroll
        for (int mt = 0; mt < 4; ++mt) a_cur[mt] = frag_ld_lds(Abuf, mt * 16, 0);
#pragma unroll
        for (int kk = 0; kk < CHUNK_; kk += 32) {
            if (kk + 32 < CHUNK_) {
                b_nxt = frag_ld(W, K, n0, kbase + kk + 32);
#pragma unroll
                for (int mt = 0; mt < 4; ++mt)
                    a_nxt[mt] = frag_ld_lds(Abuf, mt * 16, kk + 32);
            }
#pragma unroll
            for (int mt = 0; mt < 4; ++mt)
                acc[mt] = __builtin_amdgcn_wmma_f32_16x16x32_bf16(
                    false, a_cur[mt], false, b_cur, (short)0, acc[mt], false, false);
            if (kk + 32 < CHUNK_) {
#pragma unroll
                for (int mt = 0; mt < 4; ++mt) a_cur[mt] = a_nxt[mt];
                b_cur = b_nxt;
            }
        }
    }

    const float bv  = bias[n0 + (lane & 15)];
    const int   col = wave * 16 + (lane & 15);  // 0..255 = gate*64 + jl
    const int   mb  = (lane >> 4) * 8;
#pragma unroll
    for (int mt = 0; mt < 4; ++mt)
#pragma unroll
        for (int r = 0; r < 8; ++r)
            g[(mt * 16 + mb + r) * 256 + col] = acc[mt][r] + bv;
    __syncthreads();
    for (int idx = threadIdx.x; idx < 64 * 64; idx += blockDim.x) {
        const int b = idx >> 6, jl = idx & 63, j = j0 + jl;
        const float gi = g[b * 256 +       jl];
        const float gf = g[b * 256 +  64 + jl];
        const float gg = g[b * 256 + 128 + jl];
        const float go = g[b * 256 + 192 + jl];
        const float si = 1.f / (1.f + __expf(-gi));
        const float sf = 1.f / (1.f + __expf(-gf));
        const float so = 1.f / (1.f + __expf(-go));
        const float c  = sf * c_state[b * RNN_ + j] + si * tanhf(gg);
        const float h  = so * tanhf(c);
        c_state[b * RNN_ + j] = c;
        h_out[b * RNN_ + j]   = h;
        if (h_out2) h_out2[(long)b * h2_stride + j] = h;
    }
}

// ---------------------------------------------------------------------------
// Location-sensitive attention: one block per batch item, 256 threads (=T_IN).
// ---------------------------------------------------------------------------
__global__ void attention_kernel(const float* __restrict__ ah,      // [64,1024]
                                 const float* __restrict__ query_w, // [128,1024]
                                 const float* __restrict__ pm,      // [64*256,128]
                                 const float* __restrict__ memory,  // [64,256,512]
                                 const int*   __restrict__ lin_len, // [64]
                                 const float* __restrict__ v_w,     // [128]
                                 const float* __restrict__ conv_w,  // [32,2,31]
                                 const float* __restrict__ dense_w, // [128,32]
                                 float* __restrict__ aw,            // [64,256]
                                 float* __restrict__ awc,           // [64,256]
                                 float* __restrict__ ctx,           // [64,512]
                                 float* __restrict__ aligns_t) {    // + t*256
    const int b = blockIdx.x;
    const int tid = threadIdx.x;
    __shared__ float s_pq[ATT_];
    __shared__ float s_aw[T_IN_], s_awc[T_IN_];
    __shared__ float s_e[T_IN_], s_tmp[T_IN_];
    __shared__ float s_cw[NF_ * 62];
    __shared__ float s_dw[ATT_ * NF_];
    __shared__ float s_v[ATT_];

    s_aw[tid]  = aw[b * T_IN_ + tid];
    s_awc[tid] = awc[b * T_IN_ + tid];
    for (int i = tid; i < NF_ * 62;  i += 256) s_cw[i] = conv_w[i];
    for (int i = tid; i < ATT_ * NF_; i += 256) s_dw[i] = dense_w[i];
    if (tid < ATT_) s_v[tid] = v_w[tid];
    if (tid < ATT_) {                 // pq = ah @ query_w^T
        float s = 0.f;
        const float* hr = ah + b * RNN_;
        const float* qr = query_w + tid * RNN_;
        for (int k = 0; k < RNN_; ++k) s += hr[k] * qr[k];
        s_pq[tid] = s;
    }
    __syncthreads();
    const int len = lin_len[b];
    {
        const int t = tid;
        float f[NF_];
#pragma unroll
        for (int n = 0; n < NF_; ++n) f[n] = 0.f;
        for (int k = 0; k < KC_; ++k) {        // cross-correlation, PAD=15
            const int src = t + k - 15;
            const bool ok = (src >= 0) && (src < T_IN_);
            const float a0 = ok ? s_aw[src]  : 0.f;
            const float a1 = ok ? s_awc[src] : 0.f;
#pragma unroll
            for (int n = 0; n < NF_; ++n)
                f[n] += s_cw[n * 62 + k] * a0 + s_cw[n * 62 + 31 + k] * a1;
        }
        float e = -1e9f;
        if (t < len) {
            float acc = 0.f;
            const float* pr = pm + ((size_t)b * T_IN_ + t) * ATT_;
            for (int a = 0; a < ATT_; ++a) {
                float loc = 0.f;
#pragma unroll
                for (int n = 0; n < NF_; ++n) loc += s_dw[a * NF_ + n] * f[n];
                acc += s_v[a] * tanhf(s_pq[a] + loc + pr[a]);
            }
            e = acc;
        }
        s_e[t] = e;
    }
    __syncthreads();
    s_tmp[tid] = s_e[tid];
    __syncthreads();
    for (int off = 128; off > 0; off >>= 1) {   // max-reduce
        if (tid < off) s_tmp[tid] = fmaxf(s_tmp[tid], s_tmp[tid + off]);
        __syncthreads();
    }
    const float mx = s_tmp[0];
    __syncthreads();
    const float ex = __expf(s_e[tid] - mx);
    s_tmp[tid] = ex;
    __syncthreads();
    for (int off = 128; off > 0; off >>= 1) {   // sum-reduce
        if (tid < off) s_tmp[tid] += s_tmp[tid + off];
        __syncthreads();
    }
    const float awn  = ex / s_tmp[0];
    const float awcn = s_awc[tid] + awn;
    __syncthreads();
    s_aw[tid] = awn;
    aw[b * T_IN_ + tid]  = awn;
    awc[b * T_IN_ + tid] = awcn;
    aligns_t[(size_t)b * (T_MEL_ * T_IN_) + tid] = awn;
    __syncthreads();
    for (int e2 = tid; e2 < EMB_; e2 += 256) {  // ctx = aw @ memory
        float s = 0.f;
        const float* mr = memory + (size_t)b * T_IN_ * EMB_ + e2;
        for (int t2 = 0; t2 < T_IN_; ++t2) s += s_aw[t2] * mr[(size_t)t2 * EMB_];
        ctx[b * EMB_ + e2] = s;
    }
}

// ---------------------------------------------------------------------------
// Small helper kernels: conversions / packing / bias sums.
// ---------------------------------------------------------------------------
__global__ void cvt_pack2_kernel(const float* __restrict__ w0, int c0,
                                 const float* __restrict__ w1, int c1,
                                 bf16* __restrict__ dst, int rows) {
    const int  ld    = c0 + c1;
    const long total = (long)rows * ld;
    for (long i = (long)blockIdx.x * blockDim.x + threadIdx.x; i < total;
         i += (long)gridDim.x * blockDim.x) {
        const long r = i / ld;
        const int  c = (int)(i - r * ld);
        const float v = (c < c0) ? w0[r * c0 + c] : w1[r * c1 + (c - c0)];
        dst[i] = (bf16)v;
    }
}

__global__ void cvt_pad_kernel(const float* __restrict__ src, int cs,
                               bf16* __restrict__ dst, int cd, long rows) {
    const long total = rows * cd;
    for (long i = (long)blockIdx.x * blockDim.x + threadIdx.x; i < total;
         i += (long)gridDim.x * blockDim.x) {
        const long r = i / cd;
        const int  c = (int)(i - r * cd);
        dst[i] = (bf16)((c < cs) ? src[r * cs + c] : 0.f);
    }
}

__global__ void bias_sum_kernel(const float* __restrict__ a,
                                const float* __restrict__ b,
                                float* __restrict__ o, int n) {
    const int i = blockIdx.x * blockDim.x + threadIdx.x;
    if (i < n) o[i] = a[i] + b[i];
}

// A_row[row] = bf16(cat(a[row], b[row], c[row])); per-row strides in floats.
__global__ void pack3_kernel(const float* __restrict__ a, long ars, int ca,
                             const float* __restrict__ b, long brs, int cb,
                             const float* __restrict__ c, long crs, int cc,
                             bf16* __restrict__ dst, int ldd) {
    const int row = blockIdx.y;
    for (int i = blockIdx.x * blockDim.x + threadIdx.x; i < ldd;
         i += gridDim.x * blockDim.x) {
        float v;
        if (i < ca)           v = a[row * ars + i];
        else if (i < ca + cb) v = b[row * brs + (i - ca)];
        else                  v = c[row * crs + (i - ca - cb)];
        dst[(size_t)row * ldd + i] = (bf16)v;
    }
}

// ---------------------------------------------------------------------------
extern "C" void kernel_launch(void* const* d_in, const int* in_sizes, int n_in,
                              void* d_out, int out_size, void* d_ws, size_t ws_size,
                              hipStream_t stream) {
    (void)in_sizes; (void)n_in; (void)out_size; (void)ws_size;
    const float* linguistic = (const float*)d_in[0];
    const float* mel        = (const float*)d_in[1];
    const int*   lin_len    = (const int*)d_in[2];
    const float* w1         = (const float*)d_in[4];
    const float* w2         = (const float*)d_in[5];
    const float* mel_wih    = (const float*)d_in[6];
    const float* mel_whh    = (const float*)d_in[7];
    const float* mel_bih    = (const float*)d_in[8];
    const float* mel_bhh    = (const float*)d_in[9];
    const float* enc_wih    = (const float*)d_in[10];
    const float* enc_whh    = (const float*)d_in[11];
    const float* enc_bih    = (const float*)d_in[12];
    const float* enc_bhh    = (const float*)d_in[13];
    const float* query_w    = (const float*)d_in[14];
    const float* memory_w   = (const float*)d_in[15];
    const float* v_w        = (const float*)d_in[16];
    const float* conv_w     = (const float*)d_in[17];
    const float* dense_w    = (const float*)d_in[18];

    char* p = (char*)d_ws;
    auto alloc = [&](size_t bytes) -> void* {
        void* r = (void*)p;
        p += (bytes + 255) & ~(size_t)255;
        return r;
    };
    const int KMEL = FC_ + EMB_ + RNN_;   // 1792  (x_t | ctx | ah)
    const int KENC = RNN_ + EMB_ + RNN_;  // 2560  (ah | ctx | eh)

    bf16*  Wmel  = (bf16*)alloc(sizeof(bf16) * 4096UL * KMEL);
    bf16*  Wenc  = (bf16*)alloc(sizeof(bf16) * 4096UL * KENC);
    bf16*  W1p   = (bf16*)alloc(sizeof(bf16) * 256UL * 96);
    bf16*  W2b   = (bf16*)alloc(sizeof(bf16) * 256UL * 256);
    bf16*  Wmemb = (bf16*)alloc(sizeof(bf16) * 128UL * 512);
    bf16*  melp  = (bf16*)alloc(sizeof(bf16) * 32768UL * 96);
    bf16*  lingb = (bf16*)alloc(sizeof(bf16) * 16384UL * 512);
    bf16*  x1    = (bf16*)alloc(sizeof(bf16) * 32768UL * 256);
    float* xpre  = (float*)alloc(sizeof(float) * 32768UL * 256);
    float* pm    = (float*)alloc(sizeof(float) * 16384UL * 128);
    float* bias_mel = (float*)alloc(sizeof(float) * 4096);
    float* bias_enc = (float*)alloc(sizeof(float) * 4096);
    float* ah  = (float*)alloc(sizeof(float) * B_ * RNN_);
    float* ac  = (float*)alloc(sizeof(float) * B_ * RNN_);
    float* eh  = (float*)alloc(sizeof(float) * B_ * RNN_);
    float* ec  = (float*)alloc(sizeof(float) * B_ * RNN_);
    float* aw  = (float*)alloc(sizeof(float) * B_ * T_IN_);
    float* awc = (float*)alloc(sizeof(float) * B_ * T_IN_);
    float* ctx = (float*)alloc(sizeof(float) * B_ * EMB_);
    bf16*  Amel = (bf16*)alloc(sizeof(bf16) * B_ * KMEL);
    bf16*  Aenc = (bf16*)alloc(sizeof(bf16) * B_ * KENC);

    // --- one-time prep (re-done each launch; deterministic) -----------------
    cvt_pack2_kernel<<<2048, 256, 0, stream>>>(mel_wih, 768, mel_whh, 1024, Wmel, 4096);
    cvt_pack2_kernel<<<2048, 256, 0, stream>>>(enc_wih, 1536, enc_whh, 1024, Wenc, 4096);
    cvt_pad_kernel<<<96, 256, 0, stream>>>(w1, 80, W1p, 96, 256);
    cvt_pad_kernel<<<256, 256, 0, stream>>>(w2, 256, W2b, 256, 256);
    cvt_pad_kernel<<<256, 256, 0, stream>>>(memory_w, 512, Wmemb, 512, 128);
    cvt_pad_kernel<<<2048, 256, 0, stream>>>(mel, 80, melp, 96, 32768);
    cvt_pad_kernel<<<2048, 256, 0, stream>>>(linguistic, 512, lingb, 512, 16384);
    bias_sum_kernel<<<16, 256, 0, stream>>>(mel_bih, mel_bhh, bias_mel, 4096);
    bias_sum_kernel<<<16, 256, 0, stream>>>(enc_bih, enc_bhh, bias_enc, 4096);

    hipMemsetAsync(ah,  0, sizeof(float) * B_ * RNN_, stream);
    hipMemsetAsync(ac,  0, sizeof(float) * B_ * RNN_, stream);
    hipMemsetAsync(eh,  0, sizeof(float) * B_ * RNN_, stream);
    hipMemsetAsync(ec,  0, sizeof(float) * B_ * RNN_, stream);
    hipMemsetAsync(aw,  0, sizeof(float) * B_ * T_IN_, stream);
    hipMemsetAsync(awc, 0, sizeof(float) * B_ * T_IN_, stream);
    hipMemsetAsync(ctx, 0, sizeof(float) * B_ * EMB_, stream);

    // Prenet (relu(relu(mel@W1^T)@W2^T)) and processed_memory, all WMMA.
    wmma_gemm_kernel<true,  true ><<<dim3(512, 2), 256, 0, stream>>>(melp, 96,  W1p,  96,  x1,   256, 96);
    wmma_gemm_kernel<false, true ><<<dim3(512, 2), 256, 0, stream>>>(x1,   256, W2b,  256, xpre, 256, 256);
    wmma_gemm_kernel<false, false><<<dim3(256, 1), 256, 0, stream>>>(lingb, 512, Wmemb, 512, pm,  128, 512);

    float* outs   = (float*)d_out;                               // [B,T_MEL,RNN]
    float* aligns = (float*)d_out + (size_t)B_ * T_MEL_ * RNN_;  // [B,T_MEL,T_IN]

    for (int t = 0; t < T_MEL_; ++t) {
        // A_mel = bf16([x_t | ctx | ah_prev]);  x_t row b lives at (b*T_MEL+t)*FC
        pack3_kernel<<<dim3(7, 64), 256, 0, stream>>>(
            xpre + (size_t)t * FC_, (long)T_MEL_ * FC_, FC_,
            ctx, EMB_, EMB_, ah, RNN_, RNN_, Amel, KMEL);
        lstm_step_kernel<<<16, 512, 0, stream>>>(Amel, Wmel, bias_mel,
                                                 ah, ac, nullptr, 0, KMEL);
        attention_kernel<<<64, 256, 0, stream>>>(ah, query_w, pm, linguistic,
                                                 lin_len, v_w, conv_w, dense_w,
                                                 aw, awc, ctx,
                                                 aligns + (size_t)t * T_IN_);
        // A_enc = bf16([ah | ctx | eh_prev])
        pack3_kernel<<<dim3(10, 64), 256, 0, stream>>>(
            ah, RNN_, RNN_, ctx, EMB_, EMB_, eh, RNN_, RNN_, Aenc, KENC);
        lstm_step_kernel<<<16, 512, 0, stream>>>(Aenc, Wenc, bias_enc,
                                                 eh, ec,
                                                 outs + (size_t)t * RNN_,
                                                 (long)T_MEL_ * RNN_, KENC);
    }
}
